// MultiheadCoAttention_53876069761614
// MI455X (gfx1250) — matrix operand order repaired and checked by reading, not compiled, not run on hardware
//
#include <hip/hip_runtime.h>
#include <hip/hip_bf16.h>
#include <cstdint>
#include <cstddef>

// ---------------------------------------------------------------------------
// MI455X (gfx1250) multi-head co-attention, bf16 WMMA pipeline.
//  - all GEMM inputs pre-converted to bf16 (one-time pass)
//  - double-buffered tile staging via GLOBAL_LOAD_ASYNC_TO_LDS_B128 (ASYNCcnt)
//  - flash V staging split into load(regs)/store(LDS) phases so compute
//    ds_loads never queue behind the next tile's stores
//  - softmax row reductions via fused single-instruction DPP16 butterflies
// ---------------------------------------------------------------------------

#define ASYNC_STAGE 1

typedef __attribute__((ext_vector_type(16))) __bf16 v16bf;
typedef __attribute__((ext_vector_type(8)))  float  v8f;

#define DEV __device__ __forceinline__

constexpr int Bsz = 4, Lq = 1024, Dd = 1024, Aa = 1024, Hh = 16, DHd = 64;
constexpr int Mrows = Bsz * Lq;              // 4096
constexpr float SCALE = 0.03125f;            // 1/sqrt(1024)
constexpr float LN_EPS = 1e-5f;

DEV unsigned short f2bf(float f) {
    unsigned int u = __builtin_bit_cast(unsigned int, f);
    unsigned int r = u + 0x7FFFu + ((u >> 16) & 1u);   // round-to-nearest-even
    return (unsigned short)(r >> 16);
}
DEV unsigned int pack2(float a, float b) {
    return (unsigned int)f2bf(a) | ((unsigned int)f2bf(b) << 16);
}

union FragAB { v16bf v; uint4 q[2]; };

// ---- fused DPP16 butterfly reductions (1 VALU op per step) -----------------
DEV float red_max16(float x) {
    asm("v_max_num_f32_dpp %0, %0, %0 quad_perm:[1,0,3,2] row_mask:0xf bank_mask:0xf bound_ctrl:1\n\t"
        "v_max_num_f32_dpp %0, %0, %0 quad_perm:[2,3,0,1] row_mask:0xf bank_mask:0xf bound_ctrl:1\n\t"
        "v_max_num_f32_dpp %0, %0, %0 row_half_mirror row_mask:0xf bank_mask:0xf bound_ctrl:1\n\t"
        "v_max_num_f32_dpp %0, %0, %0 row_mirror row_mask:0xf bank_mask:0xf bound_ctrl:1"
        : "+v"(x));
    return x;
}
DEV float red_add16(float x) {
    asm("v_add_f32_dpp %0, %0, %0 quad_perm:[1,0,3,2] row_mask:0xf bank_mask:0xf bound_ctrl:1\n\t"
        "v_add_f32_dpp %0, %0, %0 quad_perm:[2,3,0,1] row_mask:0xf bank_mask:0xf bound_ctrl:1\n\t"
        "v_add_f32_dpp %0, %0, %0 row_half_mirror row_mask:0xf bank_mask:0xf bound_ctrl:1\n\t"
        "v_add_f32_dpp %0, %0, %0 row_mirror row_mask:0xf bank_mask:0xf bound_ctrl:1"
        : "+v"(x));
    return x;
}

// ---- async 16-byte global -> LDS copy (ASYNCcnt) ---------------------------
DEV void async_copy16(const unsigned short* gptr, unsigned short* lptr) {
#if ASYNC_STAGE
    unsigned lds = (unsigned)(uintptr_t)lptr;          // low 32 bits = LDS offset
    unsigned long long ga = (unsigned long long)gptr;
    asm volatile("global_load_async_to_lds_b128 %0, %1, off"
                 :: "v"(lds), "v"(ga) : "memory");
#else
    *(uint4*)lptr = *(const uint4*)gptr;
#endif
}
DEV void async_wait() {
#if ASYNC_STAGE
    asm volatile("s_wait_asynccnt 0x0" ::: "memory");
#endif
}

// ---------------------------------------------------------------------------
// Kernel 0: f32 -> bf16 bulk conversion (4 elements / thread).
// ---------------------------------------------------------------------------
__global__ __launch_bounds__(256)
void f32_to_bf16_kernel(const float* __restrict__ src,
                        unsigned short* __restrict__ dst, int n4)
{
    const int i = blockIdx.x * 256 + threadIdx.x;
    if (i < n4) {
        const float4 v = ((const float4*)src)[i];
        uint2 p; p.x = pack2(v.x, v.y); p.y = pack2(v.z, v.w);
        ((uint2*)dst)[i] = p;
    }
}

// ---------------------------------------------------------------------------
// GEMM: C[M,N] = A[M,K] * W[N,K]^T (+ A2*W2^T) (+ bias), K = 1024, bf16 in.
// Block tile 128x64, 8 waves of 32x32 each, f32 accumulate.
// Double-buffered async staging: tile t+1 streams while tile t computes.
// ---------------------------------------------------------------------------
constexpr int LDA = 72;                       // padded LDS stride (halves)

DEV void gemm_stage(const unsigned short* __restrict__ Ap,
                    const unsigned short* __restrict__ Wp,
                    unsigned short* As, unsigned short* Ws,
                    int Mbase, int Nbase, int k0, int srow, int scol)
{
#pragma unroll
    for (int t = 0; t < 4; ++t) {
        const int row = srow + t * 32;
        async_copy16(Ap + (size_t)(Mbase + row) * Dd + k0 + scol,
                     As + row * LDA + scol);
    }
#pragma unroll
    for (int t = 0; t < 2; ++t) {
        const int row = srow + t * 32;
        async_copy16(Wp + (size_t)(Nbase + row) * Dd + k0 + scol,
                     Ws + row * LDA + scol);
    }
}

template <bool OUT_F32>
__global__ __launch_bounds__(256)
void gemm_nt_kernel(const unsigned short* __restrict__ A,
                    const unsigned short* __restrict__ W,
                    const unsigned short* __restrict__ A2,
                    const unsigned short* __restrict__ W2,
                    const float* __restrict__ bias, void* __restrict__ Cptr)
{
    __shared__ unsigned short As[2][128 * LDA];
    __shared__ unsigned short Ws[2][64 * LDA];

    const int tid  = threadIdx.x;
    const int lane = tid & 31;
    const int wv   = tid >> 5;                    // 0..7
    const int wm   = wv >> 1;                     // 0..3 (M dir)
    const int wn   = wv & 1;                      // 0..1 (N dir)
    const int Mbase = blockIdx.y * 128;
    const int Nbase = blockIdx.x * 64;

    const int srow = tid >> 3;                    // 0..31  (staging row)
    const int scol = (tid & 7) * 8;               // 0..56  (staging col, halves)

    const int lrow   = lane & 15;
    const int half8  = (lane < 16) ? 0 : 8;       // A-fragment lane-half offset
    const int half16 = (lane < 16) ? 0 : 16;      // unused; B fragments use half8 too

    v8f acc[2][2];
#pragma unroll
    for (int i = 0; i < 2; ++i)
#pragma unroll
        for (int j = 0; j < 2; ++j)
#pragma unroll
            for (int e = 0; e < 8; ++e) acc[i][j][e] = 0.0f;

    const int T = (A2 != nullptr) ? 32 : 16;      // tiles of 64 k, over 1..2 passes

    gemm_stage(A, W, As[0], Ws[0], Mbase, Nbase, 0, srow, scol);
    async_wait();
    __syncthreads();

    for (int t = 0; t < T; ++t) {
        const int cur = t & 1;
        if (t + 1 < T) {
            const unsigned short* Ap = ((t + 1) & 16) ? A2 : A;
            const unsigned short* Wp = ((t + 1) & 16) ? W2 : W;
            gemm_stage(Ap, Wp, As[cur ^ 1], Ws[cur ^ 1],
                       Mbase, Nbase, ((t + 1) & 15) * 64, srow, scol);
        }

#pragma unroll
        for (int kk = 0; kk < 64; kk += 32) {
            FragAB af[2], bf_[2];
#pragma unroll
            for (int i = 0; i < 2; ++i) {
                const unsigned short* p = As[cur] + (wm * 32 + i * 16 + lrow) * LDA + kk + half8;
                af[i].q[0] = *(const uint4*)p;
                af[i].q[1] = *(const uint4*)(p + 16);
            }
#pragma unroll
            for (int j = 0; j < 2; ++j) {
                const unsigned short* p = Ws[cur] + (wn * 32 + j * 16 + lrow) * LDA + kk + half8;
                bf_[j].q[0] = *(const uint4*)p;
                bf_[j].q[1] = *(const uint4*)(p + 16);
            }
#pragma unroll
            for (int i = 0; i < 2; ++i)
#pragma unroll
                for (int j = 0; j < 2; ++j)
                    acc[i][j] = __builtin_amdgcn_wmma_f32_16x16x32_bf16(
                        false, af[i].v, false, bf_[j].v,
                        (short)0, acc[i][j], false, false);
        }
        async_wait();
        __syncthreads();
    }

    // ---- epilogue: C fragment -> global (row = vgpr, col = lane) ------------
#pragma unroll
    for (int i = 0; i < 2; ++i) {
#pragma unroll
        for (int j = 0; j < 2; ++j) {
            const int col = Nbase + wn * 32 + j * 16 + lrow;
            const float bv = bias ? bias[col] : 0.0f;
#pragma unroll
            for (int e = 0; e < 8; ++e) {
                const int row = Mbase + wm * 32 + i * 16 + e + half8;
                const float val = acc[i][j][e] + bv;
                if constexpr (OUT_F32)
                    ((float*)Cptr)[(size_t)row * Aa + col] = val;
                else
                    ((unsigned short*)Cptr)[(size_t)row * Aa + col] = f2bf(val);
            }
        }
    }
}

// ---------------------------------------------------------------------------
// Flash attention. Grid (B*H, L/128). 8 waves x 16 query rows.
// Double-buffered staging: K via async (phase A), V transposed from registers
// after the compute section (phase B).
// ---------------------------------------------------------------------------
constexpr int LKS = 72;   // K tile stride  (32 keys x 64 dh)
constexpr int LVS = 40;   // V^T tile stride (64 dh x 32 keys)
constexpr int LPS = 40;   // per-wave P stride (16 rows x 32 keys)

__global__ __launch_bounds__(256)
void flash_attn_kernel(const unsigned short* __restrict__ Q,
                       const unsigned short* __restrict__ K,
                       const unsigned short* __restrict__ V,
                       const int* __restrict__ mask,
                       unsigned short* __restrict__ O)
{
    __shared__ unsigned short Ksm[2][32 * LKS];
    __shared__ unsigned short Vtsm[2][64 * LVS];
    __shared__ unsigned short Psm[8 * 16 * LPS];
    __shared__ float biassm[2][32];

    const int tid  = threadIdx.x;
    const int lane = tid & 31;
    const int wv   = tid >> 5;
    const int lrow   = lane & 15;
    const int half8  = (lane < 16) ? 0 : 8;
    const int half16 = (lane < 16) ? 0 : 16;

    const int bh = blockIdx.x;
    const int b  = bh >> 4;
    const int h  = bh & 15;
    const int qrow0 = blockIdx.y * 128 + wv * 16;

    const unsigned short* Qb = Q + (size_t)(b * Lq) * Aa + h * DHd;
    const unsigned short* Kb = K + (size_t)(b * Lq) * Aa + h * DHd;
    const unsigned short* Vb = V + (size_t)(b * Lq) * Aa + h * DHd;

    const int skey = tid >> 3;        // 0..31 : key row staged by this thread
    const int scol = (tid & 7) * 8;   // dh column group (halves)

    // ---- two-phase staging for tile kt -> buffer buf ------------------------
    uint4 vreg;                        // V chunk in registers between phases
    float mreg = 1.0f;                 // mask value between phases
    auto stageA = [&](int kt, int buf) {       // async K + V/mask into regs
        const int keybase = kt * 32;
        async_copy16(Kb + (size_t)(keybase + skey) * Aa + scol,
                     Ksm[buf] + skey * LKS + scol);
        vreg = *(const uint4*)(Vb + (size_t)(keybase + skey) * Aa + scol);
        if (tid < 32) mreg = (float)mask[b * Lq + keybase + tid];
    };
    auto stageB = [&](int buf) {               // publish V (transposed) + bias
        const unsigned short* vs = (const unsigned short*)&vreg;
#pragma unroll
        for (int c = 0; c < 8; ++c)
            Vtsm[buf][(scol + c) * LVS + skey] = vs[c];
        if (tid < 32) biassm[buf][tid] = (1.0f - mreg) * -100000.0f;
    };

    // ---- Q fragments: 16 rows x 64 dh, in registers for the whole pass ------
    FragAB qf[2];
#pragma unroll
    for (int f = 0; f < 2; ++f) {
        const unsigned short* p = Qb + (size_t)(qrow0 + lrow) * Aa + f * 32 + half8;
        qf[f].q[0] = *(const uint4*)p;
        qf[f].q[1] = *(const uint4*)(p + 16);
    }

    float mrow[8], lsum[8];
    v8f oacc[4];
#pragma unroll
    for (int e = 0; e < 8; ++e) { mrow[e] = -3.0e38f; lsum[e] = 0.0f; }
#pragma unroll
    for (int n = 0; n < 4; ++n)
#pragma unroll
        for (int e = 0; e < 8; ++e) oacc[n][e] = 0.0f;

    stageA(0, 0);
    stageB(0);
    async_wait();
    __syncthreads();

    for (int kt = 0; kt < 32; ++kt) {
        const int cur = kt & 1;
        const bool more = (kt + 1 < 32);
        if (more) stageA(kt + 1, cur ^ 1);

        // ---- S = Q * K^T  (two 16x16 key sub-tiles, k-dim = dh = 64) --------
        v8f s[2];
#pragma unroll
        for (int ns = 0; ns < 2; ++ns) {
#pragma unroll
            for (int e = 0; e < 8; ++e) s[ns][e] = 0.0f;
#pragma unroll
            for (int f = 0; f < 2; ++f) {
                FragAB kb_;
                const unsigned short* p = Ksm[cur] + (ns * 16 + lrow) * LKS + f * 32 + half8;
                kb_.q[0] = *(const uint4*)p;
                kb_.q[1] = *(const uint4*)(p + 16);
                s[ns] = __builtin_amdgcn_wmma_f32_16x16x32_bf16(
                    false, qf[f].v, false, kb_.v, (short)0, s[ns], false, false);
            }
            const float bc = biassm[cur][ns * 16 + lrow];
#pragma unroll
            for (int e = 0; e < 8; ++e) s[ns][e] = s[ns][e] * SCALE + bc;
        }

        // ---- online softmax: fused DPP16 row reductions ---------------------
        unsigned short* Pw = Psm + wv * 16 * LPS;
#pragma unroll
        for (int e = 0; e < 8; ++e) {
            const float rm = red_max16(fmaxf(s[0][e], s[1][e]));
            const float mnew = fmaxf(mrow[e], rm);
            const float sc = __expf(mrow[e] - mnew);
            mrow[e] = mnew;
            const float p0 = __expf(s[0][e] - mnew);
            const float p1 = __expf(s[1][e] - mnew);
            const float rs = red_add16(p0 + p1);
            lsum[e] = lsum[e] * sc + rs;
#pragma unroll
            for (int n = 0; n < 4; ++n) oacc[n][e] *= sc;
            const int prow = e + half8;
            Pw[prow * LPS + lrow]      = f2bf(p0);
            Pw[prow * LPS + 16 + lrow] = f2bf(p1);
        }
        asm volatile("" ::: "memory");   // keep LDS P stores before reload

        // ---- O += P * V  (k-dim = 32 keys, 4 dh sub-tiles) ------------------
        FragAB pa;
        {
            const unsigned short* p = Pw + lrow * LPS + half8;
            pa.q[0] = *(const uint4*)p;
            pa.q[1] = *(const uint4*)(p + 16);
        }
#pragma unroll
        for (int n = 0; n < 4; ++n) {
            FragAB vb_;
            const unsigned short* p = Vtsm[cur] + (n * 16 + lrow) * LVS + half8;
            vb_.q[0] = *(const uint4*)p;
            vb_.q[1] = *(const uint4*)(p + 16);
            oacc[n] = __builtin_amdgcn_wmma_f32_16x16x32_bf16(
                false, pa.v, false, vb_.v, (short)0, oacc[n], false, false);
        }

        if (more) stageB(cur ^ 1);       // publish next V tile after our loads
        async_wait();
        __syncthreads();
    }

    // ---- normalize and store O (bf16, [b*L+q, h*64+dh]) ---------------------
    unsigned short* Ob = O + (size_t)(b * Lq) * Aa + h * DHd;
#pragma unroll
    for (int e = 0; e < 8; ++e) {
        const float inv = 1.0f / lsum[e];
        const int row = qrow0 + e + half8;
#pragma unroll
        for (int n = 0; n < 4; ++n)
            Ob[(size_t)row * Aa + n * 16 + lrow] = f2bf(oacc[n][e] * inv);
    }
}

// ---------------------------------------------------------------------------
// Per-row LayerNorm over 1024 features.
// ---------------------------------------------------------------------------
__global__ __launch_bounds__(256)
void layernorm_kernel(const float* __restrict__ X, const float* __restrict__ gamma,
                      const float* __restrict__ beta, float* __restrict__ Out)
{
    __shared__ float s1[8], s2[8];
    const int row  = blockIdx.x;
    const int tid  = threadIdx.x;
    const int lane = tid & 31;
    const int wv   = tid >> 5;

    const float* xr = X + (size_t)row * Aa;
    const float4 v = *(const float4*)(xr + tid * 4);
    float sum = v.x + v.y + v.z + v.w;
    float sq  = v.x * v.x + v.y * v.y + v.z * v.z + v.w * v.w;
#pragma unroll
    for (int m = 1; m < 32; m <<= 1) { sum += __shfl_xor(sum, m); sq += __shfl_xor(sq, m); }
    if (lane == 0) { s1[wv] = sum; s2[wv] = sq; }
    __syncthreads();
    if (wv == 0) {
        float a  = (lane < 8) ? s1[lane] : 0.0f;
        float bq = (lane < 8) ? s2[lane] : 0.0f;
#pragma unroll
        for (int m = 1; m < 8; m <<= 1) { a += __shfl_xor(a, m); bq += __shfl_xor(bq, m); }
        if (lane == 0) { s1[0] = a; s2[0] = bq; }
    }
    __syncthreads();
    const float mean = s1[0] * (1.0f / (float)Aa);
    const float var  = s2[0] * (1.0f / (float)Aa) - mean * mean;
    const float rstd = rsqrtf(var + LN_EPS);

    const int c0 = tid * 4;
    const float4 g  = *(const float4*)(gamma + c0);
    const float4 bt = *(const float4*)(beta + c0);
    float4 o;
    o.x = (v.x - mean) * rstd * g.x + bt.x;
    o.y = (v.y - mean) * rstd * g.y + bt.y;
    o.z = (v.z - mean) * rstd * g.z + bt.z;
    o.w = (v.w - mean) * rstd * g.w + bt.w;
    *(float4*)(Out + (size_t)row * Aa + c0) = o;
}

// ---------------------------------------------------------------------------
// Host launcher
// ---------------------------------------------------------------------------
extern "C" void kernel_launch(void* const* d_in, const int* in_sizes, int n_in,
                              void* d_out, int out_size, void* d_ws, size_t ws_size,
                              hipStream_t stream)
{
    (void)in_sizes; (void)n_in; (void)out_size; (void)ws_size;
    const float* x    = (const float*)d_in[0];
    const float* qo   = (const float*)d_in[1];
    const float* Wk   = (const float*)d_in[2];
    const float* Wqs  = (const float*)d_in[3];
    const float* Wqo  = (const float*)d_in[4];
    const float* Wv   = (const float*)d_in[5];
    const float* Wo   = (const float*)d_in[6];
    const float* bo   = (const float*)d_in[7];
    const float* gamma = (const float*)d_in[8];
    const float* beta  = (const float*)d_in[9];
    const int*   mask  = (const int*)d_in[10];

    char* ws = (char*)d_ws;
    const size_t MB = 1024 * 1024;
    unsigned short* Kbuf = (unsigned short*)(ws);               //  8 MB
    unsigned short* Qbuf = (unsigned short*)(ws +  8 * MB);     //  8 MB
    unsigned short* Vbuf = (unsigned short*)(ws + 16 * MB);     //  8 MB
    unsigned short* Obuf = (unsigned short*)(ws + 24 * MB);     //  8 MB
    float*          tmp  = (float*)        (ws + 32 * MB);      // 16 MB
    unsigned short* xb   = (unsigned short*)(ws + 48 * MB);     //  8 MB
    unsigned short* qb   = (unsigned short*)(ws + 56 * MB);     //  8 MB
    unsigned short* wkb  = (unsigned short*)(ws + 64 * MB);     //  2 MB
    unsigned short* wqsb = (unsigned short*)(ws + 66 * MB);     //  2 MB
    unsigned short* wqob = (unsigned short*)(ws + 68 * MB);     //  2 MB
    unsigned short* wvb  = (unsigned short*)(ws + 70 * MB);     //  2 MB
    unsigned short* wob  = (unsigned short*)(ws + 72 * MB);     //  2 MB

    // ---- one-time f32 -> bf16 conversions ----------------------------------
    const int nAct4 = Mrows * Aa / 4;     // 1M float4 per activation tensor
    const int nW4   = Aa * Dd / 4;        // 256K float4 per weight
    f32_to_bf16_kernel<<<dim3((nAct4 + 255) / 256), dim3(256), 0, stream>>>(x,  xb,  nAct4);
    f32_to_bf16_kernel<<<dim3((nAct4 + 255) / 256), dim3(256), 0, stream>>>(qo, qb,  nAct4);
    f32_to_bf16_kernel<<<dim3((nW4 + 255) / 256),   dim3(256), 0, stream>>>(Wk,  wkb,  nW4);
    f32_to_bf16_kernel<<<dim3((nW4 + 255) / 256),   dim3(256), 0, stream>>>(Wqs, wqsb, nW4);
    f32_to_bf16_kernel<<<dim3((nW4 + 255) / 256),   dim3(256), 0, stream>>>(Wqo, wqob, nW4);
    f32_to_bf16_kernel<<<dim3((nW4 + 255) / 256),   dim3(256), 0, stream>>>(Wv,  wvb,  nW4);
    f32_to_bf16_kernel<<<dim3((nW4 + 255) / 256),   dim3(256), 0, stream>>>(Wo,  wob,  nW4);

    const dim3 gg(Aa / 64, Mrows / 128), gb(256);
    // K = x @ Wk.T ; Qc = x @ Wqs.T + queries @ Wqo.T ; V = x @ Wv.T
    gemm_nt_kernel<false><<<gg, gb, 0, stream>>>(xb, wkb,  nullptr, nullptr, nullptr, Kbuf);
    gemm_nt_kernel<false><<<gg, gb, 0, stream>>>(xb, wqsb, qb,      wqob,    nullptr, Qbuf);
    gemm_nt_kernel<false><<<gg, gb, 0, stream>>>(xb, wvb,  nullptr, nullptr, nullptr, Vbuf);
    // softmax(Qc K^T * scale + mask) V
    flash_attn_kernel<<<dim3(Bsz * Hh, Lq / 128), gb, 0, stream>>>(Qbuf, Kbuf, Vbuf, mask, Obuf);
    // out = O @ Wo.T + bo
    gemm_nt_kernel<true><<<gg, gb, 0, stream>>>(Obuf, wob, nullptr, nullptr, bo, tmp);
    // layernorm
    layernorm_kernel<<<Mrows, gb, 0, stream>>>(tmp, gamma, beta, (float*)d_out);
}
